// MLP_Model_61649960567383
// MI455X (gfx1250) — matrix operand order repaired and checked by reading, not compiled, not run on hardware
//
#include <hip/hip_runtime.h>
#include <hip/hip_bf16.h>

typedef __attribute__((ext_vector_type(16))) _Float16 v16h;
typedef __attribute__((ext_vector_type(8)))  _Float16 v8h;
typedef __attribute__((ext_vector_type(8)))  float    v8f;
typedef __attribute__((ext_vector_type(4)))  float    v4f;

#define FDIM 1024
#define DDIM 128
#define KC   128          // K chunk staged in LDS per iteration (kernel 1)
#define P1   136          // LDS pitch (f16) for kernel-1 W^T chunk (16B-aligned rows)
#define PW1  264          // LDS pitch for Wm1^T (256 K + pad, 16B-aligned)
#define PW2  72           // LDS pitch for Wm2^T / per-wave staging tile

// ---------------------------------------------------------------------------
// Kernel 0: one-time weight prep. Convert all weights to f16, transposed to
// [n][k] so each B fragment is two contiguous 8-half runs. Outputs live in ws
// and stay L2-resident (<0.6 MB total).
// ---------------------------------------------------------------------------
__global__ __launch_bounds__(256) void prep_weights(
    const float* __restrict__ W1, const float* __restrict__ W2,
    const float* __restrict__ Wm1, const float* __restrict__ Wm2,
    _Float16* __restrict__ WT1g, _Float16* __restrict__ WT2g,
    _Float16* __restrict__ WTm1g, _Float16* __restrict__ WTm2g)
{
    const int i = blockIdx.x * 256 + threadIdx.x;
    if (i < DDIM * FDIM) {                      // [1024,128] -> [128][1024]
        const int n = i >> 10, k = i & 1023;
        WT1g[i] = (_Float16)W1[(size_t)k * DDIM + n];
        WT2g[i] = (_Float16)W2[(size_t)k * DDIM + n];
    }
    if (i < 64 * 256) {                         // [256,64] -> [64][256]
        const int n = i >> 8, k = i & 255;
        WTm1g[i] = (_Float16)Wm1[k * 64 + n];
    }
    if (i < 32 * 64) {                          // [64,32] -> [32][64]
        const int n = i >> 6, k = i & 63;
        WTm2g[i] = (_Float16)Wm2[k * 32 + n];
    }
}

// ---------------------------------------------------------------------------
// Kernel 1: h = X @ W + b for both node types, output stored as f16.
// Grid: (ceil(Nmax/128), 2). Block: 256 threads = 8 waves; wave w owns rows
// [mBase + 16w, +16), computing a 16x128 strip via 8 WMMA n-tiles.
// Staging is now a pure b128 copy of pre-converted f16 weights.
// ---------------------------------------------------------------------------
__global__ __launch_bounds__(256) void proj_kernel(
    const float* __restrict__ xg, const float* __restrict__ xc,
    const _Float16* __restrict__ WT1g, const float* __restrict__ b1,
    const _Float16* __restrict__ WT2g, const float* __restrict__ b2,
    _Float16* __restrict__ h1, _Float16* __restrict__ h2,
    int N1, int N2)
{
    __shared__ __attribute__((aligned(16))) _Float16 WT[DDIM * P1]; // 34,816 B

    const int which = blockIdx.y;
    const float*    X    = which ? xc : xg;
    const _Float16* WTg  = which ? WT2g : WT1g;
    const float*    bias = which ? b2 : b1;
    _Float16*       H    = which ? h2 : h1;
    const int       N    = which ? N2 : N1;

    const int mBase = blockIdx.x * 128;
    if (mBase >= N) return;                 // uniform per block, before barriers

    const int t    = threadIdx.x;
    const int wave = t >> 5;
    const int lane = t & 31;
    const int hi   = lane >> 4;             // which half of the wave
    const int l15  = lane & 15;

    const int row = mBase + wave * 16 + l15;
    const int rr  = row < N ? row : N - 1;  // clamp so EXEC stays all-1s
    const float* xr = X + (size_t)rr * FDIM;

    float bcol[8];
#pragma unroll
    for (int nt = 0; nt < 8; ++nt) bcol[nt] = bias[nt * 16 + l15];

    v8f acc[8] = {};

    for (int kc = 0; kc < FDIM; kc += KC) {
        __syncthreads();
        // Vector copy of pre-transposed f16 chunk: WT[n][k] = WTg[n][kc+k]
#pragma unroll
        for (int i = t * 8; i < DDIM * KC; i += 256 * 8) {
            const int n = i >> 7, kf = i & (KC - 1);
            *(v8h*)&WT[n * P1 + kf] = *(const v8h*)&WTg[(size_t)n * FDIM + kc + kf];
        }
        __syncthreads();

#pragma unroll
        for (int kk = 0; kk < KC; kk += 32) {
            // --- A fragment: lane holds row rr, K chunks {base, base+16} of 8 ---
            v16h a;
            {
                const int kg = kc + kk + hi * 8;
                const v4f f0 = *(const v4f*)(xr + kg);
                const v4f f1 = *(const v4f*)(xr + kg + 4);
                const v4f f2 = *(const v4f*)(xr + kg + 16);
                const v4f f3 = *(const v4f*)(xr + kg + 20);
#pragma unroll
                for (int j = 0; j < 4; ++j) {
                    a[j]      = (_Float16)f0[j];
                    a[4 + j]  = (_Float16)f1[j];
                    a[8 + j]  = (_Float16)f2[j];
                    a[12 + j] = (_Float16)f3[j];
                }
            }
            // --- Load ALL 8 B fragments first (one DS clause, one wait) ---
            v16h bfr[8];
#pragma unroll
            for (int nt = 0; nt < 8; ++nt) {
                const _Float16* bp = &WT[(nt * 16 + l15) * P1 + kk + hi * 8];
                const v8h b0 = *(const v8h*)bp;
                const v8h b1v = *(const v8h*)(bp + 16);
#pragma unroll
                for (int j = 0; j < 8; ++j) { bfr[nt][j] = b0[j]; bfr[nt][8 + j] = b1v[j]; }
            }
            // --- 8 independent back-to-back WMMAs ---
#pragma unroll
            for (int nt = 0; nt < 8; ++nt)
                acc[nt] = __builtin_amdgcn_wmma_f32_16x16x32_f16(
                    false, a, false, bfr[nt], (short)0, acc[nt], false, false);
        }
    }

    // Epilogue: C layout -> row = i + 8*hi, col = nt*16 + l15. Store f16.
#pragma unroll
    for (int nt = 0; nt < 8; ++nt) {
        const int col = nt * 16 + l15;
#pragma unroll
        for (int i = 0; i < 8; ++i) {
            const int r = mBase + wave * 16 + i + hi * 8;
            if (r < N) H[(size_t)r * DDIM + col] = (_Float16)(acc[nt][i] + bcol[nt]);
        }
    }
}

// ---------------------------------------------------------------------------
// Kernel 2: per-edge 3-layer MLP, persistent grid-stride over 16-edge tiles.
// Each wave owns one tile per iteration. Gathered f16 features (L2-resident)
// feed WMMA layer 1 (K=256 -> 64), LDS-transposed into WMMA layer 2
// (K=64 -> 32) whose B fragments are register-resident, then a VALU dot for
// 32 -> 1.
// ---------------------------------------------------------------------------
__global__ __launch_bounds__(256) void edge_mlp_kernel(
    const _Float16* __restrict__ h1, const _Float16* __restrict__ h2,
    const int* __restrict__ eidx,
    const _Float16* __restrict__ WTm1g, const float* __restrict__ bm1,
    const _Float16* __restrict__ WTm2g, const float* __restrict__ bm2,
    const float* __restrict__ Wm3, const float* __restrict__ bm3,
    float* __restrict__ out, int E)
{
    __shared__ __attribute__((aligned(16))) _Float16 WT1[64 * PW1];     // 33,792 B
    __shared__ __attribute__((aligned(16))) _Float16 WT2[32 * PW2];     //  4,608 B
    __shared__ __attribute__((aligned(16))) _Float16 Sv[8][16 * PW2];   // 18,432 B

    const int t = threadIdx.x;
    // Vector-copy pre-converted weights into padded LDS images.
#pragma unroll
    for (int i = t * 8; i < 64 * 256; i += 256 * 8) {
        const int n = i >> 8, kf = i & 255;
        *(v8h*)&WT1[n * PW1 + kf] = *(const v8h*)&WTm1g[i];
    }
    {
        const int i = t * 8;
        if (i < 32 * 64) {
            const int n = i >> 6, kf = i & 63;
            *(v8h*)&WT2[n * PW2 + kf] = *(const v8h*)&WTm2g[i];
        }
    }
    __syncthreads();

    const int wave = t >> 5;
    const int lane = t & 31;
    const int hi   = lane >> 4;
    const int l15  = lane & 15;
    _Float16* S = Sv[wave];

    // Tile-invariant values: layer-2 B fragments + biases, hoisted to registers.
    v16h b2f[2][2];
#pragma unroll
    for (int ks = 0; ks < 2; ++ks)
#pragma unroll
        for (int nt = 0; nt < 2; ++nt) {
            const _Float16* bp = &WT2[(nt * 16 + l15) * PW2 + ks * 32 + hi * 8];
            const v8h b0 = *(const v8h*)bp;
            const v8h b1v = *(const v8h*)(bp + 16);
#pragma unroll
            for (int j = 0; j < 8; ++j) { b2f[ks][nt][j] = b0[j]; b2f[ks][nt][8 + j] = b1v[j]; }
        }
    float bc1[4], bc2[2];
#pragma unroll
    for (int nt = 0; nt < 4; ++nt) bc1[nt] = bm1[nt * 16 + l15];
#pragma unroll
    for (int nt = 0; nt < 2; ++nt) bc2[nt] = bm2[nt * 16 + l15];
    const float b3 = bm3[0];

    const int ntiles = (E + 15) >> 4;          // E % 16 == 0: tiles fully valid
    for (int tile = blockIdx.x * 8 + wave; tile < ntiles; tile += gridDim.x * 8) {
        const int tb  = tile * 16;
        const int e   = tb + l15;
        const int src = eidx[e];
        const int dst = eidx[E + e];
        const _Float16* r1 = h1 + (size_t)src * DDIM;
        const _Float16* r2 = h2 + (size_t)dst * DDIM;

        // ---- Layer 1: [16 x 256] @ [256 x 64] ----
        v8f acc1[4] = {};
#pragma unroll
        for (int kk = 0; kk < 256; kk += 32) {
            v16h a;
#pragma unroll
            for (int c = 0; c < 2; ++c) {
                const int kb = kk + c * 16 + hi * 8;   // never straddles the 128 split
                const v8h av = (kb < 128) ? *(const v8h*)(r1 + kb)
                                          : *(const v8h*)(r2 + (kb - 128));
#pragma unroll
                for (int j = 0; j < 8; ++j) a[c * 8 + j] = av[j];
            }
            v16h bfr[4];
#pragma unroll
            for (int nt = 0; nt < 4; ++nt) {
                const _Float16* bp = &WT1[(nt * 16 + l15) * PW1 + kk + hi * 8];
                const v8h b0 = *(const v8h*)bp;
                const v8h b1v = *(const v8h*)(bp + 16);
#pragma unroll
                for (int j = 0; j < 8; ++j) { bfr[nt][j] = b0[j]; bfr[nt][8 + j] = b1v[j]; }
            }
#pragma unroll
            for (int nt = 0; nt < 4; ++nt)
                acc1[nt] = __builtin_amdgcn_wmma_f32_16x16x32_f16(
                    false, a, false, bfr[nt], (short)0, acc1[nt], false, false);
        }
        // bias + relu, transpose through per-wave LDS tile into A layout
#pragma unroll
        for (int nt = 0; nt < 4; ++nt) {
            const int col = nt * 16 + l15;
#pragma unroll
            for (int i = 0; i < 8; ++i) {
                float v = acc1[nt][i] + bc1[nt];
                S[(i + hi * 8) * PW2 + col] = (_Float16)(v > 0.f ? v : 0.f);
            }
        }

        // ---- Layer 2: [16 x 64] @ [64 x 32], B register-resident ----
        v8f acc2[2] = {};
#pragma unroll
        for (int ks = 0; ks < 2; ++ks) {
            v16h a;
#pragma unroll
            for (int c = 0; c < 2; ++c) {
                const v8h av = *(const v8h*)&S[l15 * PW2 + ks * 32 + c * 16 + hi * 8];
#pragma unroll
                for (int j = 0; j < 8; ++j) a[c * 8 + j] = av[j];
            }
#pragma unroll
            for (int nt = 0; nt < 2; ++nt)
                acc2[nt] = __builtin_amdgcn_wmma_f32_16x16x32_f16(
                    false, a, false, b2f[ks][nt], (short)0, acc2[nt], false, false);
        }
        // bias + relu; reuse S (layer-1 data already consumed)
#pragma unroll
        for (int nt = 0; nt < 2; ++nt) {
            const int col = nt * 16 + l15;
#pragma unroll
            for (int i = 0; i < 8; ++i) {
                float v = acc2[nt][i] + bc2[nt];
                S[(i + hi * 8) * PW2 + col] = (_Float16)(v > 0.f ? v : 0.f);
            }
        }

        // ---- Layer 3: [16 x 32] @ [32 x 1] on lanes 0..15 ----
        if (hi == 0) {
            float sum = b3;
#pragma unroll
            for (int j = 0; j < 32; ++j)
                sum += (float)S[l15 * PW2 + j] * Wm3[j];
            out[tb + l15] = sum;
        }
    }
}

extern "C" void kernel_launch(void* const* d_in, const int* in_sizes, int n_in,
                              void* d_out, int out_size, void* d_ws, size_t ws_size,
                              hipStream_t stream) {
    const float* xg  = (const float*)d_in[0];
    const float* xc  = (const float*)d_in[1];
    const int*   ei  = (const int*)  d_in[2];
    const float* W1  = (const float*)d_in[3];
    const float* b1  = (const float*)d_in[4];
    const float* W2  = (const float*)d_in[5];
    const float* b2  = (const float*)d_in[6];
    const float* Wm1 = (const float*)d_in[7];
    const float* bm1 = (const float*)d_in[8];
    const float* Wm2 = (const float*)d_in[9];
    const float* bm2 = (const float*)d_in[10];
    const float* Wm3 = (const float*)d_in[11];
    const float* bm3 = (const float*)d_in[12];
    float* out = (float*)d_out;

    const int N1 = in_sizes[0] / FDIM;
    const int N2 = in_sizes[1] / FDIM;
    const int E  = in_sizes[2] / 2;

    // Workspace layout (all offsets 256B-aligned)
    char* ws = (char*)d_ws;
    _Float16* h1    = (_Float16*)ws;                     ws += (size_t)N1 * DDIM * 2;
    _Float16* h2    = (_Float16*)ws;                     ws += (size_t)N2 * DDIM * 2;
    _Float16* WT1g  = (_Float16*)ws;                     ws += (size_t)DDIM * FDIM * 2;
    _Float16* WT2g  = (_Float16*)ws;                     ws += (size_t)DDIM * FDIM * 2;
    _Float16* WTm1g = (_Float16*)ws;                     ws += 64 * 256 * 2;
    _Float16* WTm2g = (_Float16*)ws;

    prep_weights<<<(DDIM * FDIM + 255) / 256, 256, 0, stream>>>(
        W1, W2, Wm1, Wm2, WT1g, WT2g, WTm1g, WTm2g);

    const int Nmax    = N1 > N2 ? N1 : N2;
    const int mblocks = (Nmax + 127) / 128;
    dim3 g1(mblocks, 2);
    proj_kernel<<<g1, 256, 0, stream>>>(xg, xc, WT1g, b1, WT2g, b2, h1, h2, N1, N2);

    const int ntiles  = (E + 15) / 16;
    int eblocks = (ntiles + 7) / 8;
    if (eblocks > 2048) eblocks = 2048;       // persistent-ish grid
    edge_mlp_kernel<<<eblocks, 256, 0, stream>>>(h1, h2, ei, WTm1g, bm1, WTm2g, bm2,
                                                 Wm3, bm3, out, E);
}
